// ModelA_ViT_52097953300544
// MI455X (gfx1250) — compile-verified, matrix operand order checked by reading, not run on hardware
//
#include <hip/hip_runtime.h>

// ---------------------------------------------------------------------------
// MI455X (gfx1250) ViT forward.
//  * GEMMs: V_WMMA_F32_16X16X32_BF16, 32x64 wave tile, 128x128 block tile,
//    W slab staged in LDS via global_load_async_to_lds_b128 (ASYNCcnt wait)
//  * flash-attention per (n, head, 16-query tile) wave: QK^T and PV as WMMA,
//    online softmax on C-fragment layout, P relayout via LDS + s_wait_dscnt
//  * elementwise: patch merge, RMSNorm, RoPE split, cvt f32->bf16
// ---------------------------------------------------------------------------

typedef __attribute__((ext_vector_type(16))) __bf16 v16bf;
typedef __attribute__((ext_vector_type(8)))  __bf16 v8bf;
typedef __attribute__((ext_vector_type(8)))  float  v8f;
typedef __attribute__((ext_vector_type(4)))  int    v4i;

#define NB     4
#define DDIM   1024
#define LTOK   1024        // Hf*Wf = 32*32
#define ROWS   (NB * LTOK) // 4096
#define HEADS  16
#define DH     64

#if defined(__has_builtin)
#if __has_builtin(__builtin_amdgcn_global_load_async_to_lds_b128)
#define HAS_ASYNC_LDS 1
#endif
#endif

__device__ __forceinline__ v8f wmma_bf16(v16bf a, v16bf b, v8f c) {
  return __builtin_amdgcn_wmma_f32_16x16x32_bf16(
      /*neg_a=*/false, a, /*neg_b=*/false, b,
      /*c_mod=*/(short)0, c, /*reuse_a=*/false, /*reuse_b=*/false);
}

// Load a 16x32 bf16 A/B fragment. Per ISA 7.12.2 (16-bit A 16x32):
// lane l: row = l%16; elems 0..7 -> K = (l/16)*8 + 0..7 ; elems 8..15 -> +16.
// Works for B = W^T when W is row-major [N][K] (lane gives N, contiguous K).
__device__ __forceinline__ v16bf load_frag16(const __bf16* base, int row0,
                                             int ldk, int k0, int lane) {
  int r  = row0 + (lane & 15);
  int kk = k0 + ((lane >> 4) << 3);
  const __bf16* p = base + (size_t)r * ldk + kk;
  v8bf lo = *(const v8bf*)p;
  v8bf hi = *(const v8bf*)(p + 16);
  v16bf f;
#pragma unroll
  for (int i = 0; i < 8; ++i) { f[i] = lo[i]; f[i + 8] = hi[i]; }
  return f;
}

__device__ __forceinline__ float gelu_tanh(float v) {
  float u = 0.7978845608028654f * (v + 0.044715f * v * v * v);
  return 0.5f * v * (1.0f + tanhf(u));
}

__device__ __forceinline__ float halfmax16(float v) {
#pragma unroll
  for (int m = 1; m < 16; m <<= 1) v = fmaxf(v, __shfl_xor(v, m, 32));
  return v;
}
__device__ __forceinline__ float halfsum16(float v) {
#pragma unroll
  for (int m = 1; m < 16; m <<= 1) v += __shfl_xor(v, m, 32);
  return v;
}

// ---------------------------------------------------------------------------
// Generic GEMM: out[m][n] = act(sum_k A[m][k]*W[n][k] + bias[n]) (+ residual)
// 256 threads = 8 waves (4 m-waves x 2 n-waves), 32x64 per wave -> 128x128.
// W slab [128 x 32k] staged in LDS per K-step (async-to-LDS if available);
// B fragments read from LDS (shared by 4 m-waves), A fragments from global.
// ---------------------------------------------------------------------------
__global__ __launch_bounds__(256) void gemm_bf16_kernel(
    const __bf16* __restrict__ A, const __bf16* __restrict__ W,
    const float* __restrict__ bias, const float* __restrict__ residual,
    float* __restrict__ outF, __bf16* __restrict__ outB,
    int M, int N, int K, int doGelu) {
  __shared__ __bf16 wtile[128 * 32];  // [n_local][k] row-major, 8KB
  int lane = threadIdx.x & 31;
  int wave = threadIdx.x >> 5;
  int m0   = blockIdx.x * 128 + (wave & 3) * 32;
  int nblk = blockIdx.y * 128;
  int n0l  = (wave >> 2) * 64;        // local n offset of this wave's 64 cols

  v8f acc[2][4] = {};
  for (int k0 = 0; k0 < K; k0 += 32) {
    __syncthreads();  // previous iteration's ds reads done
    // cooperative copy of W[nblk..nblk+127][k0..k0+31] -> wtile
#pragma unroll
    for (int it = 0; it < 2; ++it) {
      int ch = threadIdx.x + 256 * it;   // 512 chunks of 16B
      int rw = ch >> 2;
      int q  = (ch & 3) * 8;
      const __bf16* src = W + (size_t)(nblk + rw) * K + k0 + q;
      __bf16* dst = &wtile[rw * 32 + q];
#ifdef HAS_ASYNC_LDS
      __builtin_amdgcn_global_load_async_to_lds_b128(
          (v4i*)src,
          (__attribute__((address_space(3))) v4i*)dst, 0, 0);
#else
      *(v8bf*)dst = *(const v8bf*)src;
#endif
    }
#ifdef HAS_ASYNC_LDS
    asm volatile("s_wait_asynccnt 0" ::: "memory");
#endif
    __syncthreads();

    if (k0 + 32 < K)  // gfx1250 prefetch hint for next A slab
      __builtin_prefetch((const void*)(A + (size_t)(m0 + (lane & 15)) * K + k0 + 32), 0, 1);

    v16bf a[2], b[4];
    a[0] = load_frag16(A, m0,      K, k0, lane);
    a[1] = load_frag16(A, m0 + 16, K, k0, lane);
#pragma unroll
    for (int ni = 0; ni < 4; ++ni)
      b[ni] = load_frag16(wtile, n0l + ni * 16, 32, 0, lane);
#pragma unroll
    for (int mi = 0; mi < 2; ++mi)
#pragma unroll
      for (int ni = 0; ni < 4; ++ni)
        acc[mi][ni] = wmma_bf16(a[mi], b[ni], acc[mi][ni]);
  }

  // C/D layout: lane l, vgpr r -> row = r + 8*(l/16), col = l%16
#pragma unroll
  for (int mi = 0; mi < 2; ++mi)
#pragma unroll
    for (int ni = 0; ni < 4; ++ni)
#pragma unroll
      for (int r = 0; r < 8; ++r) {
        int row = m0 + mi * 16 + r + ((lane >> 4) << 3);
        int col = nblk + n0l + ni * 16 + (lane & 15);
        float v = acc[mi][ni][r];
        if (bias) v += bias[col];
        if (doGelu) v = gelu_tanh(v);
        size_t o = (size_t)row * N + col;
        if (residual) v += residual[o];
        if (outF) outF[o] = v;
        if (outB) outB[o] = (__bf16)v;
      }
}

// ---------------------------------------------------------------------------
// Flash attention: one wave per (n, head, 16-query tile). Key tiles of 32.
// Q,K: [N][H][L][DH] bf16 (post-RoPE). V: [N][H][DH][L] bf16 (transposed).
// Output: [N][L][D] bf16 (= [row][h*64+dh]).
// ---------------------------------------------------------------------------
__global__ __launch_bounds__(256) void flash_attn_kernel(
    const __bf16* __restrict__ Q, const __bf16* __restrict__ Km,
    const __bf16* __restrict__ Vt, __bf16* __restrict__ O) {
  __shared__ __bf16 plds[8][16 * 32];  // per-wave P staging tile
  int lane = threadIdx.x & 31;
  int wv   = threadIdx.x >> 5;
  int w    = blockIdx.x * 8 + wv;      // 0..4095
  int qt = w & 63;
  int h  = (w >> 6) & 15;
  int n  = w >> 10;

  const __bf16* Qh = Q  + (((size_t)(n * HEADS + h)) * LTOK + qt * 16) * DH;
  const __bf16* Kh = Km + ((size_t)(n * HEADS + h)) * LTOK * DH;
  const __bf16* Vh = Vt + ((size_t)(n * HEADS + h)) * DH * LTOK;

  v16bf q0 = load_frag16(Qh, 0, DH, 0, lane);
  v16bf q1 = load_frag16(Qh, 0, DH, 32, lane);

  float mrow[8], lrow[8];
  v8f oacc[4] = {};
#pragma unroll
  for (int r = 0; r < 8; ++r) { mrow[r] = -1e30f; lrow[r] = 0.f; }

  const float scale = 0.125f;  // 1/sqrt(64)
  for (int s0 = 0; s0 < LTOK; s0 += 32) {
    v8f s[2];
#pragma unroll
    for (int st = 0; st < 2; ++st) {
      v16bf kb0 = load_frag16(Kh, s0 + st * 16, DH, 0, lane);
      v16bf kb1 = load_frag16(Kh, s0 + st * 16, DH, 32, lane);
      v8f a = {};
      a = wmma_bf16(q0, kb0, a);
      a = wmma_bf16(q1, kb1, a);
      s[st] = a * scale;
    }
    float alpha[8];
#pragma unroll
    for (int r = 0; r < 8; ++r) {
      float mx = halfmax16(fmaxf(s[0][r], s[1][r]));
      float mnew = fmaxf(mrow[r], mx);
      float p0 = __expf(s[0][r] - mnew);
      float p1 = __expf(s[1][r] - mnew);
      float rs = halfsum16(p0 + p1);
      alpha[r] = __expf(mrow[r] - mnew);
      lrow[r] = lrow[r] * alpha[r] + rs;
      mrow[r] = mnew;
      int row = r + ((lane >> 4) << 3);
      plds[wv][row * 32 + (lane & 15)]      = (__bf16)p0;
      plds[wv][row * 32 + 16 + (lane & 15)] = (__bf16)p1;
    }
#pragma unroll
    for (int j = 0; j < 4; ++j)
#pragma unroll
      for (int r = 0; r < 8; ++r) oacc[j][r] *= alpha[r];
    asm volatile("s_wait_dscnt 0" ::: "memory");  // LDS stores visible to reload
    v16bf pf = load_frag16(&plds[wv][0], 0, 32, 0, lane);
#pragma unroll
    for (int j = 0; j < 4; ++j) {
      v16bf vf = load_frag16(Vh, j * 16, LTOK, s0, lane);  // B: col=dh, K=s
      oacc[j] = wmma_bf16(pf, vf, oacc[j]);
    }
  }

#pragma unroll
  for (int j = 0; j < 4; ++j)
#pragma unroll
    for (int r = 0; r < 8; ++r) {
      int t   = qt * 16 + r + ((lane >> 4) << 3);
      int col = h * DH + j * 16 + (lane & 15);
      float v = oacc[j][r] / lrow[r];
      O[((size_t)n * LTOK + t) * DDIM + col] = (__bf16)v;
    }
}

// ---------------------------------------------------------------------------
// Elementwise kernels
// ---------------------------------------------------------------------------
__global__ void cvt_kernel(const float* __restrict__ s, __bf16* __restrict__ d, int n) {
  int i = blockIdx.x * 256 + threadIdx.x;
  if (i < n) d[i] = (__bf16)s[i];
}

// x[4,16,64,64] -> patches bf16 [4096][64] with channel order (c, kh, kw)
__global__ void patches_kernel(const float* __restrict__ x, __bf16* __restrict__ out) {
  int idx = blockIdx.x * 256 + threadIdx.x;  // < 4096*64
  int c64 = idx & 63;
  int row = idx >> 6;
  int n = row >> 10, l = row & 1023;
  int c = c64 >> 2, kh = (c64 >> 1) & 1, kw = c64 & 1;
  int ph = l >> 5, pw = l & 31;
  float v = x[(((size_t)n * 16 + c) * 64 + (ph * 2 + kh)) * 64 + (pw * 2 + kw)];
  out[(size_t)row * 64 + c64] = (__bf16)v;
}

// RMSNorm per row of [4096][1024]; bf16 row-major out and/or f32 transposed out
__global__ __launch_bounds__(256) void rmsnorm_kernel(
    const float* __restrict__ in, __bf16* __restrict__ outB,
    float* __restrict__ outT) {
  __shared__ float red[256];
  int row = blockIdx.x;
  const float* xr = in + (size_t)row * DDIM;
  float s = 0.f;
  for (int i = threadIdx.x; i < DDIM; i += 256) { float v = xr[i]; s += v * v; }
  red[threadIdx.x] = s;
  __syncthreads();
  for (int st = 128; st > 0; st >>= 1) {
    if (threadIdx.x < st) red[threadIdx.x] += red[threadIdx.x + st];
    __syncthreads();
  }
  float r = rsqrtf(red[0] * (1.f / DDIM) + 1e-6f);
  for (int i = threadIdx.x; i < DDIM; i += 256) {
    float v = xr[i] * r;
    if (outB) outB[(size_t)row * DDIM + i] = (__bf16)v;
    if (outT) {
      int n = row >> 10, l = row & 1023;
      outT[((size_t)n << 20) + ((size_t)i << 10) + l] = v;  // [N][D][L]
    }
  }
}

// qkv f32 [4096][3072] -> 2D-RoPE'd Q,K bf16 [N][H][L][64], V^T bf16 [N][H][64][L]
__global__ void rope_split_kernel(const float* __restrict__ qkv,
                                  __bf16* __restrict__ qh,
                                  __bf16* __restrict__ kh,
                                  __bf16* __restrict__ vt) {
  int idx = blockIdx.x * 256 + threadIdx.x;  // < 4096*16*32
  int pair = idx & 31;
  int h    = (idx >> 5) & 15;
  int row  = idx >> 9;          // n*1024 + l
  int n = row >> 10, l = row & 1023;
  int g = pair >> 4, i = pair & 15;         // g=0: pos_h, g=1: pos_w; half=16
  float pos = (g == 0) ? (float)(l >> 5) : (float)(l & 31);
  float ang = pos * __powf(10000.f, -(float)i * (1.f / 16.f));
  float sn, cs;
  __sincosf(ang, &sn, &cs);
  int base = h * DH + g * 32 + i;
  const float* qr = qkv + (size_t)row * 3072;
  float q0 = qr[base],        q1 = qr[base + 16];
  float k0 = qr[1024 + base], k1 = qr[1024 + base + 16];
  size_t hd = (((size_t)(n * HEADS + h)) * LTOK + l) * DH + g * 32 + i;
  qh[hd]      = (__bf16)(q0 * cs - q1 * sn);
  qh[hd + 16] = (__bf16)(q1 * cs + q0 * sn);
  kh[hd]      = (__bf16)(k0 * cs - k1 * sn);
  kh[hd + 16] = (__bf16)(k1 * cs + k0 * sn);
  size_t vd = (((size_t)(n * HEADS + h)) * DH + g * 32 + i) * LTOK + l;
  vt[vd]             = (__bf16)qr[2048 + base];
  vt[vd + 16 * LTOK] = (__bf16)qr[2048 + base + 16];
}

// ---------------------------------------------------------------------------
extern "C" void kernel_launch(void* const* d_in, const int* in_sizes, int n_in,
                              void* d_out, int out_size, void* d_ws, size_t ws_size,
                              hipStream_t stream) {
  const float* x       = (const float*)d_in[0];
  const float* merge_w = (const float*)d_in[1];
  const float* qkv_w   = (const float*)d_in[2];
  const float* qkv_b   = (const float*)d_in[3];
  const float* proj_w  = (const float*)d_in[4];
  const float* proj_b  = (const float*)d_in[5];
  const float* fc1_w   = (const float*)d_in[6];
  const float* fc1_b   = (const float*)d_in[7];
  const float* fc2_w   = (const float*)d_in[8];
  const float* fc2_b   = (const float*)d_in[9];

  char* p = (char*)d_ws;
  auto alloc = [&](size_t bytes) -> char* {
    char* r = p; p += (bytes + 255) & ~(size_t)255; return r;
  };
  __bf16* wmerge = (__bf16*)alloc((size_t)1024 * 64 * 2);
  __bf16* wqkv   = (__bf16*)alloc((size_t)8 * 3072 * 1024 * 2);
  __bf16* wproj  = (__bf16*)alloc((size_t)8 * 1024 * 1024 * 2);
  __bf16* wfc1   = (__bf16*)alloc((size_t)8 * 4096 * 1024 * 2);
  __bf16* wfc2   = (__bf16*)alloc((size_t)8 * 4096 * 1024 * 2);
  __bf16* patches = (__bf16*)alloc((size_t)ROWS * 64 * 2);
  float*  tok     = (float*) alloc((size_t)ROWS * DDIM * 4);
  __bf16* hbf     = (__bf16*)alloc((size_t)ROWS * DDIM * 2);
  float*  qkvbuf  = (float*) alloc((size_t)ROWS * 3072 * 4);
  __bf16* qh      = (__bf16*)alloc((size_t)ROWS * DDIM * 2);
  __bf16* khb     = (__bf16*)alloc((size_t)ROWS * DDIM * 2);
  __bf16* vtb     = (__bf16*)alloc((size_t)ROWS * DDIM * 2);
  __bf16* attnO   = (__bf16*)alloc((size_t)ROWS * DDIM * 2);
  __bf16* hidden  = (__bf16*)alloc((size_t)ROWS * 4096 * 2);

  auto cvt = [&](const float* s, __bf16* d, int n) {
    cvt_kernel<<<(n + 255) / 256, 256, 0, stream>>>(s, d, n);
  };
  cvt(merge_w, wmerge, 1024 * 64);
  cvt(qkv_w,   wqkv,   8 * 3072 * 1024);
  cvt(proj_w,  wproj,  8 * 1024 * 1024);
  cvt(fc1_w,   wfc1,   8 * 4096 * 1024);
  cvt(fc2_w,   wfc2,   8 * 4096 * 1024);

  patches_kernel<<<(ROWS * 64) / 256, 256, 0, stream>>>(x, patches);

  // tok = patches @ merge_w^T   (M=4096, N=1024, K=64)
  gemm_bf16_kernel<<<dim3(ROWS / 128, 1024 / 128), 256, 0, stream>>>(
      patches, wmerge, nullptr, nullptr, tok, nullptr, ROWS, 1024, 64, 0);

  for (int i = 0; i < 8; ++i) {
    rmsnorm_kernel<<<ROWS, 256, 0, stream>>>(tok, hbf, nullptr);
    // qkv = h @ qkv_w^T + b   (M=4096, N=3072, K=1024)
    gemm_bf16_kernel<<<dim3(ROWS / 128, 3072 / 128), 256, 0, stream>>>(
        hbf, wqkv + (size_t)i * 3072 * 1024, qkv_b + (size_t)i * 3072,
        nullptr, qkvbuf, nullptr, ROWS, 3072, 1024, 0);
    rope_split_kernel<<<(ROWS * HEADS * 32) / 256, 256, 0, stream>>>(
        qkvbuf, qh, khb, vtb);
    flash_attn_kernel<<<(NB * HEADS * 64) / 8, 256, 0, stream>>>(
        qh, khb, vtb, attnO);
    // tok += attnO @ proj_w^T + b   (M=4096, N=1024, K=1024)
    gemm_bf16_kernel<<<dim3(ROWS / 128, 1024 / 128), 256, 0, stream>>>(
        attnO, wproj + (size_t)i * 1024 * 1024, proj_b + (size_t)i * 1024,
        tok, tok, nullptr, ROWS, 1024, 1024, 0);

    rmsnorm_kernel<<<ROWS, 256, 0, stream>>>(tok, hbf, nullptr);
    // hidden = gelu(h @ fc1_w^T + b)   (M=4096, N=4096, K=1024)
    gemm_bf16_kernel<<<dim3(ROWS / 128, 4096 / 128), 256, 0, stream>>>(
        hbf, wfc1 + (size_t)i * 4096 * 1024, fc1_b + (size_t)i * 4096,
        nullptr, nullptr, hidden, ROWS, 4096, 1024, 1);
    // tok += hidden @ fc2_w^T + b   (M=4096, N=1024, K=4096)
    gemm_bf16_kernel<<<dim3(ROWS / 128, 1024 / 128), 256, 0, stream>>>(
        hidden, wfc2 + (size_t)i * 4096 * 1024, fc2_b + (size_t)i * 1024,
        tok, tok, nullptr, ROWS, 1024, 4096, 0);
  }

  // final RMSNorm, transposed write into d_out [N][D][L]
  rmsnorm_kernel<<<ROWS, 256, 0, stream>>>(tok, nullptr, (float*)d_out);
}
